// ConvNet_STDP_65386582114561
// MI455X (gfx1250) — matrix operand order repaired and checked by reading, not compiled, not run on hardware
//
#include <hip/hip_runtime.h>

typedef __attribute__((ext_vector_type(16))) _Float16 v16h;
typedef __attribute__((ext_vector_type(8)))  _Float16 v8h;
typedef __attribute__((ext_vector_type(8)))  float    v8f;

// ---------------- problem dimensions ----------------
#define BATCH 32
#define T_STEPS 20
// layer 1
#define H0 240
#define W0 160
#define C1 4
#define H1 236
#define W1c 156
#define HW1 (H1*W1c)               // 36816
#define SZ1 ((size_t)BATCH*C1*HW1) // 4,712,448
// pool1 (k=7,s=6)
#define P1H 39
#define P1W 25
#define P1 (P1H*P1W)               // 975
#define SZP1 ((size_t)BATCH*C1*P1) // 124,800
// layer 2
#define C2 20
#define H2 24
#define W2c 10
#define N2 (H2*W2c)                // 240 valid positions
#define N2P 256                    // padded to 16 N-tiles
#define K2 1024                    // 4*16*16
#define SZ2 ((size_t)BATCH*C2*N2)  // 153,600
// pool2 (k=2,s=2)
#define P2 60                      // 12*5
#define SZP2 ((size_t)BATCH*C2*P2) // 38,400
// layer 3
#define C3 10
#define H3 8
#define SZ3 ((size_t)BATCH*C3*H3)  // 2,560

// ---------------- workspace layout (float units; byte regions carved out) ---
#define O_V1   ((size_t)0)
#define O_M1B  (O_V1 + SZ1)        // SZ1 bytes = SZ1/4 floats (mask1, {0,1})
#define O_Z2B  (O_M1B + SZ1/4)     // SZ1 bytes (layer-1 spikes, {0,1})
#define O_V2   (O_Z2B + SZ1/4)
#define O_M2   (O_V2 + SZ2)
#define O_V3   (O_M2 + SZ2)
#define O_M1P  (O_V3 + SZ3)
#define O_M2P  (O_M1P + SZP1)
#define O_Z3   (O_M2P + SZP2)
#define O_Z6   (O_Z3 + SZP1)
#define O_W2H  (O_Z6 + SZP2)       // 32*1024 halves (zero-padded rows 20..31)

// ---------------- init: reset state, build padded f16 weights ----------------
__global__ __launch_bounds__(256) void snn_init_kernel(float* __restrict__ ws,
                                                       const float* __restrict__ w2) {
  size_t idx = (size_t)blockIdx.x * 256 + threadIdx.x;
  unsigned char* m1b = (unsigned char*)(ws + O_M1B);
  if (idx < SZ1)  { ws[O_V1 + idx] = 0.f; m1b[idx] = 1; }
  if (idx < SZ2)  { ws[O_V2 + idx] = 0.f; ws[O_M2 + idx] = 1.f; }
  if (idx < SZ3)  { ws[O_V3 + idx] = 0.f; }
  if (idx < SZP1) { ws[O_M1P + idx] = 1.f; }
  if (idx < SZP2) { ws[O_M2P + idx] = 1.f; }
  if (idx < 32 * 1024) {
    _Float16* w2h = (_Float16*)(ws + O_W2H);
    int m = (int)(idx >> 10), k = (int)(idx & 1023);
    w2h[idx] = (m < C2) ? (_Float16)w2[m * 1024 + k] : (_Float16)0.f;
  }
}

// ---------------- layer 1: conv1 + IAF + lateral inhibition (fused) ----------
// one thread owns one (b,h,w) pixel across all 4 channels
__global__ __launch_bounds__(256) void conv1_iaf_kernel(const float* __restrict__ xt,
                                                        const float* __restrict__ w1,
                                                        float* __restrict__ v1,
                                                        unsigned char* __restrict__ m1,
                                                        unsigned char* __restrict__ z2) {
  __shared__ float w1s[C1 * 25];
  const int tid = threadIdx.x;
  for (int i = tid; i < C1 * 25; i += 256) w1s[i] = w1[i];
  __syncthreads();

  const size_t idx = (size_t)blockIdx.x * 256 + tid;
  if (idx >= (size_t)BATCH * HW1) return;
  const int b = (int)(idx / HW1);
  const int r = (int)(idx % HW1);
  const int h = r / W1c, w = r % W1c;

  const float* xb = xt + (size_t)b * (H0 * W0);
  float z[C1] = {0.f, 0.f, 0.f, 0.f};
  for (int kh = 0; kh < 5; ++kh) {
    const float* row = xb + (size_t)(h + kh) * W0 + w;
#pragma unroll
    for (int kw = 0; kw < 5; ++kw) {
      const float xv = row[kw];
#pragma unroll
      for (int c = 0; c < C1; ++c) z[c] = fmaf(xv, w1s[c * 25 + kh * 5 + kw], z[c]);
    }
  }

  const size_t base = (size_t)b * C1 * HW1 + (size_t)h * W1c + w;
  float vv[C1];
  bool sp[C1], ml[C1];
  float best = -1e30f;
  int bi = -1;
  bool any = false;
#pragma unroll
  for (int c = 0; c < C1; ++c) {
    const size_t o = base + (size_t)c * HW1;
    const float v = v1[o] + z[c];
    const bool mo = (m1[o] != 0);
    const bool s = (v >= 10.0f);
    vv[c] = v; sp[c] = s; ml[c] = mo;
    if (s && mo) { any = true; if (v > best) { best = v; bi = c; } }
  }
#pragma unroll
  for (int c = 0; c < C1; ++c) {
    const size_t o = base + (size_t)c * HW1;
    const bool nm = any ? (ml[c] && (c == bi)) : ml[c];
    const float vr = sp[c] ? (vv[c] - 10.0f) : vv[c];
    m1[o] = nm ? 1 : 0;
    v1[o] = nm ? vr : 0.f;
    z2[o] = (any && c == bi) ? 1 : 0;  // spike survives only for winner
  }
}

// ---------------- spiking max-pool 7x7 stride 6 (OR over {0,1} bytes) -------
__global__ __launch_bounds__(256) void pool1_kernel(const unsigned char* __restrict__ z2,
                                                    float* __restrict__ m1p,
                                                    float* __restrict__ z3) {
  const size_t idx = (size_t)blockIdx.x * 256 + threadIdx.x;
  if (idx >= SZP1) return;
  const int b = (int)(idx / (C1 * P1));
  int r = (int)(idx % (C1 * P1));
  const int c = r / P1; r %= P1;
  const int ph = r / P1W, pw = r % P1W;
  const unsigned char* src = z2 + (((size_t)b * C1 + c) * H1 + ph * 6) * W1c + pw * 6;
  unsigned int mx = 0;
  for (int i = 0; i < 7; ++i) {
    const unsigned char* rr = src + (size_t)i * W1c;
#pragma unroll
    for (int j = 0; j < 7; ++j) mx |= rr[j];
  }
  const float mp = m1p[idx];
  const float out = (float)mx * mp;
  m1p[idx] = mp * (1.f - (out > 0.f ? 1.f : 0.f));
  z3[idx] = out;
}

// ---------------- layer 2: im2col WMMA conv + IAF + lateral + pool2 ---------
// one block per batch element; 8 waves x 4 (Mtile,Ntile) tiles = all 32 tiles
// of the padded 32x256 output. Branch-free k-loop: global A loads, ds_load B + v_wmma.
__global__ __launch_bounds__(256) void conv2_wmma_kernel(const float* __restrict__ z3g,
                                                         const _Float16* __restrict__ w2h,
                                                         float* __restrict__ v2,
                                                         float* __restrict__ m2,
                                                         float* __restrict__ m2p,
                                                         float* __restrict__ z6) {
  __shared__ _Float16 s_z3[4096];          // 3900 spikes as f16 + zero tail
  __shared__ float s_z4[32 * N2P];         // padded 32x256 conv out / z5 spikes
  const int b = blockIdx.x;
  const int tid = threadIdx.x;

  for (int i = tid; i < C1 * P1; i += 256)
    s_z3[i] = (_Float16)z3g[(size_t)b * (C1 * P1) + i];
  for (int i = C1 * P1 + tid; i < 4096; i += 256)
    s_z3[i] = (_Float16)0.f;               // pad reads for discarded cols
  __syncthreads();

  const int wave = tid >> 5, lane = tid & 31;
  const int mt = wave & 1, ntb = wave >> 1;   // ntb in 0..3
  const int hi = lane >> 4, lo = lane & 15;
  const int m = mt * 16 + lo;
  const _Float16* wrow = w2h + (size_t)m * K2;

  // per-tile B base offsets: n_i -> oh*25 + ow inside a (c,kh) plane
  int nbase[4], ncol[4];
#pragma unroll
  for (int i = 0; i < 4; ++i) {
    const int n = (ntb + 4 * i) * 16 + lo;     // 0..255 (>=240 discarded later)
    const int oh = n / W2c, ow = n - oh * W2c;
    ncol[i] = n;
    nbase[i] = oh * P1W + ow;
  }

  const v8f zero8 = {0.f, 0.f, 0.f, 0.f, 0.f, 0.f, 0.f, 0.f};
  v8f acc0 = zero8, acc1 = zero8, acc2 = zero8, acc3 = zero8;

#pragma unroll 2
  for (int kk = 0; kk < K2 / 32; ++kk) {
    // A fragment: contiguous K order (k = hi*16 + s) -> two contiguous 8-half loads
    v16h a;
    const v8h A0 = *(const v8h*)(wrow + kk * 32 + hi * 16);
    const v8h A1 = *(const v8h*)(wrow + kk * 32 + hi * 16 + 8);
#pragma unroll
    for (int j = 0; j < 8; ++j) { a[j] = A0[j]; a[8 + j] = A1[j]; }

    // B fragment: this lane's 16 K-values are one (c,kh) row, kw=0..15
    const int k0 = kk * 32 + hi * 16;
    const int koffs = (k0 >> 8) * P1 + ((k0 >> 4) & 15) * P1W;  // c*975 + kh*25

    v16h bf;
#pragma unroll
    for (int j = 0; j < 16; ++j) bf[j] = s_z3[koffs + nbase[0] + j];
    acc0 = __builtin_amdgcn_wmma_f32_16x16x32_f16(false, a, false, bf, (short)0, acc0, false, false);
#pragma unroll
    for (int j = 0; j < 16; ++j) bf[j] = s_z3[koffs + nbase[1] + j];
    acc1 = __builtin_amdgcn_wmma_f32_16x16x32_f16(false, a, false, bf, (short)0, acc1, false, false);
#pragma unroll
    for (int j = 0; j < 16; ++j) bf[j] = s_z3[koffs + nbase[2] + j];
    acc2 = __builtin_amdgcn_wmma_f32_16x16x32_f16(false, a, false, bf, (short)0, acc2, false, false);
#pragma unroll
    for (int j = 0; j < 16; ++j) bf[j] = s_z3[koffs + nbase[3] + j];
    acc3 = __builtin_amdgcn_wmma_f32_16x16x32_f16(false, a, false, bf, (short)0, acc3, false, false);
  }

  // unconditional scatter (32-bit C/D layout: VGPR r -> M = mt*16 + hi*8 + r)
  const int mrow = mt * 16 + hi * 8;
#pragma unroll
  for (int r = 0; r < 8; ++r) {
    s_z4[(mrow + r) * N2P + ncol[0]] = acc0[r];
    s_z4[(mrow + r) * N2P + ncol[1]] = acc1[r];
    s_z4[(mrow + r) * N2P + ncol[2]] = acc2[r];
    s_z4[(mrow + r) * N2P + ncol[3]] = acc3[r];
  }
  __syncthreads();

  // IAF (v_th=60) + lateral inhibition across 20 channels, per position
  if (tid < N2) {
    const int n = tid;
    float* v2b = v2 + (size_t)b * C2 * N2;
    float* m2b = m2 + (size_t)b * C2 * N2;
    float vv[C2], ml[C2];
    bool sp[C2];
    float best = -1e30f;
    int bi = -1;
    bool any = false;
#pragma unroll
    for (int c = 0; c < C2; ++c) {
      const float v = v2b[c * N2 + n] + s_z4[c * N2P + n];
      const float mo = m2b[c * N2 + n];
      const bool s = (v >= 60.0f);
      vv[c] = v; sp[c] = s; ml[c] = mo;
      if (s && mo > 0.f) { any = true; if (v > best) { best = v; bi = c; } }
    }
#pragma unroll
    for (int c = 0; c < C2; ++c) {
      const float nm = any ? (ml[c] * ((c == bi) ? 1.f : 0.f)) : ml[c];
      const float vr = sp[c] ? (vv[c] - 60.0f) : vv[c];
      m2b[c * N2 + n] = nm;
      v2b[c * N2 + n] = nm * vr;
      s_z4[c * N2P + n] = (any && c == bi) ? 1.f : 0.f;  // z5 spikes in place
    }
  }
  __syncthreads();

  // spiking max-pool 2x2 stride 2 -> z6 (20x12x5)
  for (int j = tid; j < C2 * P2; j += 256) {
    const int c = j / P2;
    const int r = j % P2;
    const int ph = r / 5, pw = r % 5;
    const int n00 = (2 * ph) * W2c + 2 * pw;
    const float mx = fmaxf(fmaxf(s_z4[c * N2P + n00], s_z4[c * N2P + n00 + 1]),
                           fmaxf(s_z4[c * N2P + n00 + W2c], s_z4[c * N2P + n00 + W2c + 1]));
    const size_t go = (size_t)b * C2 * P2 + j;
    const float mp = m2p[go];
    const float out = mx * mp;
    m2p[go] = mp * (1.f - (out > 0.f ? 1.f : 0.f));
    z6[go] = out;
  }
}

// ---------------- layer 3: conv3 + IAF(v_th=inf) + spatial max --------------
__global__ __launch_bounds__(96) void conv3_kernel(const float* __restrict__ z6,
                                                   const float* __restrict__ w3,
                                                   float* __restrict__ v3,
                                                   float* __restrict__ out) {
  __shared__ float s_w3[C3 * C2 * 25];  // 5000
  __shared__ float s_z6[C2 * P2];       // 1200
  __shared__ float s_v[C3 * H3];        // 80
  const int b = blockIdx.x, tid = threadIdx.x;
  for (int i = tid; i < C3 * C2 * 25; i += 96) s_w3[i] = w3[i];
  for (int i = tid; i < C2 * P2; i += 96) s_z6[i] = z6[(size_t)b * C2 * P2 + i];
  __syncthreads();

  if (tid < C3 * H3) {
    const int oc = tid >> 3, oh = tid & 7;
    float acc = 0.f;
    for (int ic = 0; ic < C2; ++ic)
      for (int kh = 0; kh < 5; ++kh)
#pragma unroll
        for (int kw = 0; kw < 5; ++kw)
          acc = fmaf(s_z6[ic * P2 + (oh + kh) * 5 + kw],
                     s_w3[(oc * C2 + ic) * 25 + kh * 5 + kw], acc);
    // v_th = inf: never spikes, mask stays 1, membrane just accumulates
    const float v = v3[(size_t)b * (C3 * H3) + tid] + acc;
    v3[(size_t)b * (C3 * H3) + tid] = v;
    s_v[tid] = v;
  }
  __syncthreads();
  if (tid < C3) {
    float mx = s_v[tid * 8];
#pragma unroll
    for (int i = 1; i < 8; ++i) mx = fmaxf(mx, s_v[tid * 8 + i]);
    out[(size_t)b * C3 + tid] = mx;  // overwritten each t; t=19 value survives
  }
}

// ---------------- host-side launch sequence ---------------------------------
extern "C" void kernel_launch(void* const* d_in, const int* in_sizes, int n_in,
                              void* d_out, int out_size, void* d_ws, size_t ws_size,
                              hipStream_t stream) {
  const float* x  = (const float*)d_in[0];  // (20,32,1,240,160)
  const float* w1 = (const float*)d_in[1];  // (4,1,5,5)
  const float* w2 = (const float*)d_in[2];  // (20,4,16,16)
  const float* w3 = (const float*)d_in[3];  // (10,20,5,5)
  float* out = (float*)d_out;               // [320 z9 | 100 w1 | 20480 w2 | 5000 w3]
  float* ws = (float*)d_ws;

  unsigned char* m1b = (unsigned char*)(ws + O_M1B);
  unsigned char* z2b = (unsigned char*)(ws + O_Z2B);

  // reset state + build f16 padded weights (runs every call: deterministic)
  snn_init_kernel<<<(unsigned)(SZ1 / 256), 256, 0, stream>>>(ws, w2);

  // tuple outputs 2..4 are the weights, passed through
  hipMemcpyAsync(out + 320,   w1, 100   * sizeof(float), hipMemcpyDeviceToDevice, stream);
  hipMemcpyAsync(out + 420,   w2, 20480 * sizeof(float), hipMemcpyDeviceToDevice, stream);
  hipMemcpyAsync(out + 20900, w3, 5000  * sizeof(float), hipMemcpyDeviceToDevice, stream);

  for (int t = 0; t < T_STEPS; ++t) {
    const float* xt = x + (size_t)t * BATCH * H0 * W0;
    conv1_iaf_kernel<<<(unsigned)((BATCH * (size_t)HW1) / 256), 256, 0, stream>>>(
        xt, w1, ws + O_V1, m1b, z2b);
    pool1_kernel<<<(unsigned)((SZP1 + 255) / 256), 256, 0, stream>>>(
        z2b, ws + O_M1P, ws + O_Z3);
    conv2_wmma_kernel<<<BATCH, 256, 0, stream>>>(
        ws + O_Z3, (const _Float16*)(ws + O_W2H),
        ws + O_V2, ws + O_M2, ws + O_M2P, ws + O_Z6);
    conv3_kernel<<<BATCH, 96, 0, stream>>>(ws + O_Z6, w3, ws + O_V3, out);
  }
}
